// SSIMLoss_12489764896875
// MI455X (gfx1250) — matrix operand order repaired and checked by reading, not compiled, not run on hardware
//
#include <hip/hip_runtime.h>

typedef __attribute__((ext_vector_type(16))) _Float16 v16h;
typedef __attribute__((ext_vector_type(8)))  float    v8f;

#define IMG      512
#define WIN      11
#define HALO     5
#define TILE     16
#define REG      26          // TILE + WIN - 1
#define SSTRIDE  34          // padded row stride (halfs) to avoid bank conflicts
#define WAVES_PB 8
#define SSIM_C1  0.0001f     // 0.01^2
#define SSIM_C2  0.0009f     // 0.03^2

__device__ __forceinline__ void wave_lds_fence() {
    __builtin_amdgcn_wave_barrier();
    asm volatile("s_wait_dscnt 0" ::: "memory");
    __builtin_amdgcn_wave_barrier();
}

__device__ __forceinline__ v8f wmma16(v16h a, v16h b, v8f c) {
    // D = A(16x32 f16) * B(32x16 f16) + C(16x16 f32)
    return __builtin_amdgcn_wmma_f32_16x16x32_f16(
        /*neg_a=*/false, a, /*neg_b=*/false, b,
        /*c_mod=*/(short)0, c, /*reuse_a=*/false, /*reuse_b=*/false);
}

// Full separable 11-tap conv of one 26x26 (padded) f16 signal tile -> 16x16 f32.
// sig: REG rows x SSTRIDE halfs (cols 26..31 zeroed).  hl: REG x SSTRIDE scratch.
__device__ __forceinline__ v8f conv_sig(const _Float16* __restrict__ sig,
                                        _Float16* __restrict__ hl,
                                        v16h bwH, v16h awV, int lane) {
    v8f zero = {0.f,0.f,0.f,0.f,0.f,0.f,0.f,0.f};
    const int M  = lane & 15;
    const int hh = lane >> 4;

    // Horizontal pass: two row-chunks (rows 0..15 and 10..25) of H = S x Wh
#pragma unroll
    for (int chunk = 0; chunk < 2; ++chunk) {
        // A-matrix gather per ISA layout: lane M, K = 16*(e>>3) + 8*hh + (e&7)
        const _Float16* rp = sig + (chunk * 10 + M) * SSTRIDE + 8 * hh;
        v16h a;
#pragma unroll
        for (int v = 0; v < 8; ++v) {
            const int K0 = 16 * (v >> 2) + 2 * (v & 3);
            a[2 * v]     = rp[K0];
            a[2 * v + 1] = rp[K0 + 1];
        }
        v8f d = wmma16(a, bwH, zero);
        // Scatter D (lane holds col N=M, rows e+8*hh) into H rows (chunk offset 10)
#pragma unroll
        for (int e = 0; e < 8; ++e) {
            const int row = e + 8 * hh + chunk * 10;   // overlap rows rewritten w/ equal values
            hl[row * SSTRIDE + M] = (_Float16)d[e];
        }
    }
    wave_lds_fence();

    // Vertical pass: O = Wv x Hc.  B-layout: lane col N = M, K per A-layout.
    v16h b;
#pragma unroll
    for (int e = 0; e < 16; ++e) {
        const int K = 16 * (e >> 3) + 8 * hh + (e & 7);
        b[e] = (K < REG) ? hl[K * SSTRIDE + M] : (_Float16)0.f;
    }
    v8f o = wmma16(awV, b, zero);
    wave_lds_fence();   // order b-reads before the next signal's H writes
    return o;
}

__global__ __launch_bounds__(256)
void ssim_tile_kernel(const float* __restrict__ pred,
                      const float* __restrict__ targ,
                      const float* __restrict__ win2d,
                      float* __restrict__ partials) {
    __shared__ _Float16 wtab[16];
    __shared__ _Float16 sigP[WAVES_PB][REG * SSTRIDE];
    __shared__ _Float16 sigT[WAVES_PB][REG * SSTRIDE];
    __shared__ _Float16 sigQ[WAVES_PB][REG * SSTRIDE];
    __shared__ _Float16 Hbuf[WAVES_PB][REG * SSTRIDE];
    __shared__ float    wpart[WAVES_PB];

    const int tid  = threadIdx.x;
    const int wave = tid >> 5;
    const int lane = tid & 31;

    // 1-D Gaussian from the 2-D window's row sums (w2d = g (outer) g, sum(g)=1)
    if (tid < 16) {
        float s = 0.f;
        if (tid < WIN) {
#pragma unroll
            for (int j = 0; j < WIN; ++j) s += win2d[tid * WIN + j];
        }
        wtab[tid] = (_Float16)s;
    }
    __syncthreads();

    const int M  = lane & 15;
    const int hh = lane >> 4;

    // Banded weight operands (built once, shared by all 5 convs)
    v16h bwH;   // B: Wh[k][n] = w[k-n], lane col n = M, K per A-layout
#pragma unroll
    for (int e = 0; e < 16; ++e) {
        const int K = 16 * (e >> 3) + 8 * hh + (e & 7);
        const int t = K - M;
        bwH[e] = (t >= 0 && t < WIN) ? wtab[t] : (_Float16)0.f;
    }
    v16h awV;   // A: Wv[m][k] = w[k-m], lane row M, K per A-layout
#pragma unroll
    for (int e = 0; e < 16; ++e) {
        const int K = 16 * (e >> 3) + 8 * hh + (e & 7);
        const int t = K - M;
        awV[e] = (t >= 0 && t < WIN) ? wtab[t] : (_Float16)0.f;
    }

    // Tile coordinates: 1024 tiles (32x32) per 512x512 plane, 48 planes
    const int waveTile = blockIdx.x * WAVES_PB + wave;
    const int plane = waveTile >> 10;
    const int tix   = waveTile & 1023;
    const int r0 = (tix >> 5) << 4;
    const int c0 = (tix & 31) << 4;
    const size_t base = (size_t)plane * (IMG * IMG);

    _Float16* sp = sigP[wave];
    _Float16* st = sigT[wave];
    _Float16* sq = sigQ[wave];
    _Float16* hl = Hbuf[wave];

    // Stage 26x26 halo region as f16 (zero padding outside image; cols 26..31 zero)
    for (int r = 0; r < REG; ++r) {
        const int gr = r0 - HALO + r;
        const int gc = c0 - HALO + lane;
        float u = 0.f, v = 0.f;
        if (lane < REG && (unsigned)gr < IMG && (unsigned)gc < IMG) {
            const size_t off = base + (size_t)gr * IMG + gc;
            u = pred[off];
            v = targ[off];
        }
        sp[r * SSTRIDE + lane] = (_Float16)u;
        st[r * SSTRIDE + lane] = (_Float16)v;
    }
    wave_lds_fence();

    const v8f o_mu1 = conv_sig(sp, hl, bwH, awV, lane);
    const v8f o_mu2 = conv_sig(st, hl, bwH, awV, lane);

    for (int r = 0; r < REG; ++r) {
        const _Float16 x = sp[r * SSTRIDE + lane];
        sq[r * SSTRIDE + lane] = x * x;
    }
    wave_lds_fence();
    const v8f o_pp = conv_sig(sq, hl, bwH, awV, lane);

    for (int r = 0; r < REG; ++r) {
        const _Float16 x = st[r * SSTRIDE + lane];
        sq[r * SSTRIDE + lane] = x * x;
    }
    wave_lds_fence();
    const v8f o_tt = conv_sig(sq, hl, bwH, awV, lane);

    for (int r = 0; r < REG; ++r) {
        sq[r * SSTRIDE + lane] = sp[r * SSTRIDE + lane] * st[r * SSTRIDE + lane];
    }
    wave_lds_fence();
    const v8f o_pt = conv_sig(sq, hl, bwH, awV, lane);

    // SSIM map + per-lane partial sum over this lane's 8 output pixels
    float acc = 0.f;
#pragma unroll
    for (int e = 0; e < 8; ++e) {
        const float mu1 = o_mu1[e], mu2 = o_mu2[e];
        const float m11 = mu1 * mu1, m22 = mu2 * mu2, m12 = mu1 * mu2;
        const float s1  = o_pp[e] - m11;
        const float s2  = o_tt[e] - m22;
        const float s12 = o_pt[e] - m12;
        const float num = (2.f * m12 + SSIM_C1) * (2.f * s12 + SSIM_C2);
        const float den = (m11 + m22 + SSIM_C1) * (s1 + s2 + SSIM_C2);
        acc += num / den;
    }
#pragma unroll
    for (int d = 16; d >= 1; d >>= 1) acc += __shfl_xor(acc, d, 32);

    if (lane == 0) wpart[wave] = acc;
    __syncthreads();
    if (tid == 0) {
        float s = 0.f;
#pragma unroll
        for (int w = 0; w < WAVES_PB; ++w) s += wpart[w];
        partials[blockIdx.x] = s;
    }
}

__global__ __launch_bounds__(256)
void ssim_reduce_kernel(const float* __restrict__ partials, int n,
                        float* __restrict__ out, float invN) {
    __shared__ float smem[256];
    float a = 0.f;
    for (int i = threadIdx.x; i < n; i += 256) a += partials[i];
    smem[threadIdx.x] = a;
    __syncthreads();
    for (int k = 128; k > 0; k >>= 1) {
        if (threadIdx.x < k) smem[threadIdx.x] += smem[threadIdx.x + k];
        __syncthreads();
    }
    if (threadIdx.x == 0) out[0] = 1.0f - smem[0] * invN;
}

extern "C" void kernel_launch(void* const* d_in, const int* in_sizes, int n_in,
                              void* d_out, int out_size, void* d_ws, size_t ws_size,
                              hipStream_t stream) {
    const float* pred = (const float*)d_in[0];
    const float* targ = (const float*)d_in[1];
    const float* win  = (const float*)d_in[2];
    float* out      = (float*)d_out;
    float* partials = (float*)d_ws;

    const int nTiles = 16 * 3 * 32 * 32;      // 49152 tiles of 16x16
    const int blocks = nTiles / WAVES_PB;     // 6144 blocks, 8 waves each
    ssim_tile_kernel<<<blocks, 256, 0, stream>>>(pred, targ, win, partials);

    const float invN = 1.0f / (float)(16 * 3 * 512 * 512);
    ssim_reduce_kernel<<<1, 256, 0, stream>>>(partials, blocks, out, invN);
}
